// Model_51616916963598
// MI455X (gfx1250) — matrix operand order repaired and checked by reading, compiled, not run on hardware
//
#include <hip/hip_runtime.h>
#include <hip/hip_bf16.h>

// ---------------------------------------------------------------------------
// Cross-attention (B=8, L=1024, S=1000, H=8, E=128, d_model=1024, d_llm=4096)
// bf16 WMMA pipeline for gfx1250 (wave32, v_wmma_f32_16x16x32_bf16).
// ---------------------------------------------------------------------------

typedef __bf16 bf16_t;
typedef __attribute__((ext_vector_type(16))) __bf16 v16bf;
typedef __attribute__((ext_vector_type(8)))  __bf16 v8bf;
typedef __attribute__((ext_vector_type(8)))  float  v8f;

constexpr int kB   = 8;
constexpr int kL   = 1024;
constexpr int kS   = 1000;   // real source length
constexpr int kSP  = 1024;   // padded source length
constexpr int kH   = 8;
constexpr int kE   = 128;
constexpr int kHD  = 1024;   // H*E
constexpr int kDM  = 1024;   // d_model
constexpr int kDL  = 4096;   // d_llm

static __device__ __forceinline__ bf16_t f2bf(float x) { return (bf16_t)x; }

static __device__ __forceinline__ v8f vzero8() {
  v8f z;
#pragma unroll
  for (int i = 0; i < 8; ++i) z[i] = 0.0f;
  return z;
}

// ---- fragment loaders -----------------------------------------------------
// A matrix 16x32 bf16: lanes 0-15 row M=lane (K 0-15),
// lanes 16-31 row M=lane-16 (K 16-31).
static __device__ __forceinline__ v16bf load_a_frag(const bf16_t* __restrict__ A,
                                                    int lda, int m0, int k0, int lane) {
  int r  = lane & 15;
  int kh = lane >> 4;
  const bf16_t* p = A + (long)(m0 + r) * lda + k0 + kh * 16;
  v8bf lo = *(const v8bf*)p;          // K = kh*16 + 0..7
  v8bf hi = *(const v8bf*)(p + 8);    // K = kh*16 + 8..15
  v16bf a;
#pragma unroll
  for (int i = 0; i < 8; ++i) { a[i] = lo[i]; a[i + 8] = hi[i]; }
  return a;
}

// B matrix 32x16 bf16 with B(k,n) = W[(n0+n)*ldb + k0 + k] (i.e. W stored [N,K]
// row-major -> "TN" GEMM). Lane n=lane&15, lanes 0-15 hold K=0..15, 16-31 K=16..31.
static __device__ __forceinline__ v16bf load_bt_frag(const bf16_t* __restrict__ W,
                                                     int ldb, int n0, int k0, int lane) {
  int n  = lane & 15;
  int kh = lane >> 4;
  const bf16_t* p = W + (long)(n0 + n) * ldb + k0 + kh * 16;
  return *(const v16bf*)p;            // 32 contiguous bytes
}

static __device__ __forceinline__ v8f wmma_bf16(v16bf a, v16bf b, v8f c) {
  return __builtin_amdgcn_wmma_f32_16x16x32_bf16(false, a, false, b,
                                                 (short)0, c, false, false);
}

// ---- elementwise helpers --------------------------------------------------
__global__ void cvt_f32_bf16(const float* __restrict__ src, bf16_t* __restrict__ dst, long n) {
  long i = (long)blockIdx.x * blockDim.x + threadIdx.x;
  long stride = (long)gridDim.x * blockDim.x;
  for (; i < n; i += stride) dst[i] = f2bf(src[i]);
}

__global__ void cvt_pad_f32_bf16(const float* __restrict__ src, bf16_t* __restrict__ dst,
                                 int rows_src, int rows_dst, int cols) {
  long n = (long)rows_dst * cols;
  long i = (long)blockIdx.x * blockDim.x + threadIdx.x;
  long stride = (long)gridDim.x * blockDim.x;
  for (; i < n; i += stride) {
    int row = (int)(i / cols);
    dst[i] = (row < rows_src) ? f2bf(src[i]) : (bf16_t)0.0f;
  }
}

// dst[c*R + r] = src[r*C + c]
__global__ void transpose_bf16(const bf16_t* __restrict__ src, bf16_t* __restrict__ dst,
                               int R, int C) {
  long n = (long)R * C;
  long i = (long)blockIdx.x * blockDim.x + threadIdx.x;
  long stride = (long)gridDim.x * blockDim.x;
  for (; i < n; i += stride) {
    int r = (int)(i / C);
    int c = (int)(i % C);
    dst[(long)c * R + r] = src[i];
  }
}

// ---- generic TN GEMM: C[M,N] = A[M,K] * W[N,K]^T + bias --------------------
// 128 threads = 4 waves; block tile 128(M) x 64(N); wave tile 32 x 64
// (2 A-frags x 4 B-frags -> 8 WMMAs per 32-K step, 1.5 b128 loads per WMMA).
// Requires M%128==0, N%64==0, K%32==0.
template <bool OUTF32>
__global__ __launch_bounds__(128, 1)
void gemm_bt_wmma(const bf16_t* __restrict__ A, const bf16_t* __restrict__ W,
                  const float* __restrict__ bias, void* __restrict__ Cout,
                  int K, int lda, int ldb, int ldc) {
  int lane = threadIdx.x & 31;
  int wid  = threadIdx.x >> 5;
  int m0 = blockIdx.y * 128 + wid * 32;
  int n0 = blockIdx.x * 64;

  v8f acc[2][4];
#pragma unroll
  for (int mt = 0; mt < 2; ++mt)
#pragma unroll
    for (int nt = 0; nt < 4; ++nt) acc[mt][nt] = vzero8();

  for (int k0 = 0; k0 < K; k0 += 32) {
    v16bf a0 = load_a_frag(A, lda, m0,      k0, lane);
    v16bf a1 = load_a_frag(A, lda, m0 + 16, k0, lane);
#pragma unroll
    for (int nt = 0; nt < 4; ++nt) {
      v16bf b = load_bt_frag(W, ldb, n0 + nt * 16, k0, lane);
      acc[0][nt] = wmma_bf16(a0, b, acc[0][nt]);
      acc[1][nt] = wmma_bf16(a1, b, acc[1][nt]);
    }
  }

  int half = lane >> 4;
#pragma unroll
  for (int mt = 0; mt < 2; ++mt)
#pragma unroll
    for (int nt = 0; nt < 4; ++nt) {
      int col = n0 + nt * 16 + (lane & 15);
      float bv = bias ? bias[col] : 0.0f;
#pragma unroll
      for (int v = 0; v < 8; ++v) {
        int row = m0 + mt * 16 + v + half * 8;  // C layout: lanes 0-15 M=v, 16-31 M=v+8
        float valf = acc[mt][nt][v] + bv;
        if (OUTF32) ((float*)Cout)[(long)row * ldc + col] = valf;
        else        ((bf16_t*)Cout)[(long)row * ldc + col] = f2bf(valf);
      }
    }
}

// ---- fused attention (flash-style, online softmax) -------------------------
// grid = (L/64, B*H), block = 128 (4 independent waves, 16 query rows each).
// launch_bounds(128,1): keep Q frags + 16x128 f32 accum fully in VGPRs (no spill).
__global__ __launch_bounds__(128, 1)
void flash_attn_kernel(const bf16_t* __restrict__ Q,    // [B*L, HD] bf16
                       const bf16_t* __restrict__ Kp,   // [SP, HD]  bf16
                       const bf16_t* __restrict__ Vt,   // [HD, SP]  bf16 (V proj, transposed)
                       bf16_t* __restrict__ Out) {      // [B*L, HD] bf16
  __shared__ __align__(32) bf16_t ptile[4][16][40];   // per-wave P staging (80B rows)

  int lane = threadIdx.x & 31;
  int wid  = threadIdx.x >> 5;
  int bh = blockIdx.y;
  int b = bh >> 3;
  int h = bh & 7;
  int l0 = blockIdx.x * 64 + wid * 16;
  const float scale = 0.08838834764831845f;           // 1/sqrt(128)

  // Q fragments for this wave's 16 rows, K-dim = E in 4 chunks of 32
  const bf16_t* Qbase = Q + ((long)(b * kL + l0)) * kHD + h * kE;
  v16bf qf[4];
#pragma unroll
  for (int ec = 0; ec < 4; ++ec) qf[ec] = load_a_frag(Qbase, kHD, 0, ec * 32, lane);

  v8f o[8];                                           // 16 x 128 fp32 output accum
#pragma unroll
  for (int i = 0; i < 8; ++i) o[i] = vzero8();
  float m[8], lsum[8];
#pragma unroll
  for (int v = 0; v < 8; ++v) { m[v] = -1e30f; lsum[v] = 0.0f; }

  for (int s0 = 0; s0 < kSP; s0 += 32) {
    // scores tile 16(L) x 32(S), as two 16-column WMMA outputs
    v8f sc[2];
#pragma unroll
    for (int cb = 0; cb < 2; ++cb) {
      v8f acc = vzero8();
#pragma unroll
      for (int ec = 0; ec < 4; ++ec) {
        v16bf bfrag = load_bt_frag(Kp, kHD, s0 + cb * 16, h * kE + ec * 32, lane);
        acc = wmma_bf16(qf[ec], bfrag, acc);
      }
      int col = s0 + cb * 16 + (lane & 15);
#pragma unroll
      for (int v = 0; v < 8; ++v)
        acc[v] = (col < kS) ? acc[v] * scale : -1e30f;  // mask padded source rows
      sc[cb] = acc;
    }

    // online softmax: row = v + (lane>>4)*8; 16 lanes of the group hold the columns
#pragma unroll
    for (int v = 0; v < 8; ++v) {
      float mx = fmaxf(sc[0][v], sc[1][v]);
#pragma unroll
      for (int off = 1; off < 16; off <<= 1)
        mx = fmaxf(mx, __shfl_xor(mx, off, 32));
      float mn = fmaxf(m[v], mx);
      float corr = __expf(m[v] - mn);
      float p0 = __expf(sc[0][v] - mn);
      float p1 = __expf(sc[1][v] - mn);
      sc[0][v] = p0; sc[1][v] = p1;
      float rs = p0 + p1;
#pragma unroll
      for (int off = 1; off < 16; off <<= 1)
        rs += __shfl_xor(rs, off, 32);
      lsum[v] = lsum[v] * corr + rs;
      m[v] = mn;
#pragma unroll
      for (int f = 0; f < 8; ++f) o[f][v] *= corr;      // rescale running output
    }

    // C-layout -> A-layout transpose of P through this wave's LDS slice.
    int half = lane >> 4, nn = lane & 15;
#pragma unroll
    for (int cb = 0; cb < 2; ++cb)
#pragma unroll
      for (int v = 0; v < 8; ++v)
        ptile[wid][v + half * 8][cb * 16 + nn] = f2bf(sc[cb][v]);

    // same-wave DS ops are in-order; force the ordering + drain DScnt
    asm volatile("s_wait_dscnt 0" ::: "memory");

    v16bf pa;
    {
      int r = lane & 15, kh = lane >> 4;
#pragma unroll
      for (int j = 0; j < 8; ++j) {
        pa[j]     = ptile[wid][r][kh * 16 + j];
        pa[j + 8] = ptile[wid][r][kh * 16 + 8 + j];
      }
    }

    // O += P (16x32) * V (32x128); B(k,n) = Vt[(h*E + n)*SP + s0 + k]
#pragma unroll
    for (int nf = 0; nf < 8; ++nf) {
      v16bf bfrag = load_bt_frag(Vt, kSP, h * kE + nf * 16, s0, lane);
      o[nf] = wmma_bf16(pa, bfrag, o[nf]);
    }
  }

  // normalize and store bf16 into the per-head slice of the attn buffer
  int half = lane >> 4, nn = lane & 15;
#pragma unroll
  for (int nf = 0; nf < 8; ++nf)
#pragma unroll
    for (int v = 0; v < 8; ++v) {
      int row = b * kL + l0 + v + half * 8;
      float valf = o[nf][v] / lsum[v];
      Out[(long)row * kHD + h * kE + nf * 16 + nn] = f2bf(valf);
    }
}

// ---------------------------------------------------------------------------
extern "C" void kernel_launch(void* const* d_in, const int* in_sizes, int n_in,
                              void* d_out, int out_size, void* d_ws, size_t ws_size,
                              hipStream_t stream) {
  (void)in_sizes; (void)n_in; (void)out_size; (void)ws_size;

  const float* tgt = (const float*)d_in[0];   // [8,1024,1024]
  const float* src = (const float*)d_in[1];   // [1000,4096]
  const float* val = (const float*)d_in[2];   // [1000,4096]
  const float* Wq  = (const float*)d_in[3];   // [1024,1024]
  const float* bq  = (const float*)d_in[4];
  const float* Wk  = (const float*)d_in[5];   // [1024,4096]
  const float* bk  = (const float*)d_in[6];
  const float* Wv  = (const float*)d_in[7];   // [1024,4096]
  const float* bv  = (const float*)d_in[8];
  const float* Wo  = (const float*)d_in[9];   // [4096,1024]
  const float* bo  = (const float*)d_in[10];
  float* out = (float*)d_out;                 // [8,1024,4096]

  char* ws = (char*)d_ws;
  size_t off = 0;
  auto alloc = [&](size_t elems) -> bf16_t* {
    bf16_t* p = (bf16_t*)(ws + off);
    off = (off + elems * sizeof(bf16_t) + 255) & ~(size_t)255;
    return p;
  };
  bf16_t* Xbf   = alloc((size_t)kB * kL * kDM);   // 16 MB
  bf16_t* Wqbf  = alloc((size_t)kHD * kDM);       //  2 MB
  bf16_t* Wkbf  = alloc((size_t)kHD * kDL);       //  8 MB
  bf16_t* Wvbf  = alloc((size_t)kHD * kDL);       //  8 MB
  bf16_t* Wobf  = alloc((size_t)kDL * kHD);       //  8 MB
  bf16_t* Sbf   = alloc((size_t)kSP * kDL);       //  8 MB (zero-padded rows)
  bf16_t* Vinbf = alloc((size_t)kSP * kDL);       //  8 MB (zero-padded rows)
  bf16_t* Qbf   = alloc((size_t)kB * kL * kHD);   // 16 MB
  bf16_t* Kpbf  = alloc((size_t)kSP * kHD);       //  2 MB
  bf16_t* Vpbf  = alloc((size_t)kSP * kHD);       //  2 MB
  bf16_t* Vtbf  = alloc((size_t)kHD * kSP);       //  2 MB
  bf16_t* Attn  = alloc((size_t)kB * kL * kHD);   // 16 MB  (total ~96 MB)

  // 1) fp32 -> bf16 staging
  cvt_f32_bf16<<<2048, 256, 0, stream>>>(tgt, Xbf, (long)kB * kL * kDM);
  cvt_f32_bf16<<<512,  256, 0, stream>>>(Wq,  Wqbf, (long)kHD * kDM);
  cvt_f32_bf16<<<1024, 256, 0, stream>>>(Wk,  Wkbf, (long)kHD * kDL);
  cvt_f32_bf16<<<1024, 256, 0, stream>>>(Wv,  Wvbf, (long)kHD * kDL);
  cvt_f32_bf16<<<1024, 256, 0, stream>>>(Wo,  Wobf, (long)kDL * kHD);
  cvt_pad_f32_bf16<<<1024, 256, 0, stream>>>(src, Sbf,   kS, kSP, kDL);
  cvt_pad_f32_bf16<<<1024, 256, 0, stream>>>(val, Vinbf, kS, kSP, kDL);

  // 2) projections (WMMA GEMMs), outputs bf16
  gemm_bt_wmma<false><<<dim3(kHD / 64, (kB * kL) / 128), 128, 0, stream>>>(
      Xbf, Wqbf, bq, Qbf, kDM, kDM, kDM, kHD);
  gemm_bt_wmma<false><<<dim3(kHD / 64, kSP / 128), 128, 0, stream>>>(
      Sbf, Wkbf, bk, Kpbf, kDL, kDL, kDL, kHD);
  gemm_bt_wmma<false><<<dim3(kHD / 64, kSP / 128), 128, 0, stream>>>(
      Vinbf, Wvbf, bv, Vpbf, kDL, kDL, kDL, kHD);

  // 3) transpose V projection so P@V B-fragments load contiguously
  transpose_bf16<<<1024, 256, 0, stream>>>(Vpbf, Vtbf, kSP, kHD);

  // 4) fused attention (softmax never touches HBM)
  flash_attn_kernel<<<dim3(kL / 64, kB * kH), 128, 0, stream>>>(Qbf, Kpbf, Vtbf, Attn);

  // 5) output projection, fp32 result
  gemm_bt_wmma<true><<<dim3(kDL / 64, (kB * kL) / 128), 128, 0, stream>>>(
      Attn, Wobf, bo, out, kHD, kHD, kHD, kDL);
}
